// MoELayer_86165633892838
// MI455X (gfx1250) — matrix operand order repaired and checked by reading, not compiled, not run on hardware
//
#include <hip/hip_runtime.h>

#define NUM_EXPERTS 8
#define DMODEL 1024
#define HDIM 4096
#define TTOK 8192            // B*S = 4*2048
#define TM 128
#define TN 128
#define TK 32
#define LDP 40               // padded LDS row stride in halves (80B = 5*16B: b128-aligned, conflict-free frags)

typedef __attribute__((ext_vector_type(16))) _Float16 v16h;
typedef __attribute__((ext_vector_type(8)))  _Float16 v8h;
typedef __attribute__((ext_vector_type(8)))  float    v8f;

// ---------- WMMA fragment loaders (wave32, 16x16x32 f16) ----------
// A tile in LDS: row-major [m][k], row stride LDP halves.
// a[0..15] -> K = hi*16 + q  (one 32B region)
__device__ inline v16h load_a_frag(const _Float16* __restrict__ As, int lane) {
    const int m = lane & 15, hi = lane >> 4;
    const _Float16* p = As + m * LDP + hi * 16;
    v16h a;
#pragma unroll
    for (int q = 0; q < 8; ++q) {
        a[q]     = p[q];
        a[8 + q] = p[8 + q];
    }
    return a;
}

// B tile in LDS TRANSPOSED: [n][k]; b[i] = B[k = hj*16 + i][n] (one 32B region)
__device__ inline v16h load_b_frag(const _Float16* __restrict__ BsT, int lane) {
    const int n = lane & 15, hj = lane >> 4;
    const _Float16* p = BsT + n * LDP + hj * 16;
    v16h b;
#pragma unroll
    for (int q = 0; q < 16; ++q) b[q] = p[q];
    return b;
}

__device__ inline v8f wmma_f16(v16h a, v16h b, v8f c) {
    return __builtin_amdgcn_wmma_f32_16x16x32_f16(false, a, false, b, (short)0, c, false, false);
}

// ---------- Kernel 1: gating (one wave per token) ----------
__global__ __launch_bounds__(256) void gate_kernel(
    const float* __restrict__ x, const float* __restrict__ gw,
    const float* __restrict__ gb, int* __restrict__ counts,
    int* __restrict__ tok_id, float* __restrict__ tok_w) {
    const int lane = threadIdx.x & 31;
    const int wave = threadIdx.x >> 5;
    const int t = blockIdx.x * 8 + wave;

    float acc[NUM_EXPERTS];
#pragma unroll
    for (int e = 0; e < NUM_EXPERTS; ++e) acc[e] = 0.0f;

    const float* xr = x + (size_t)t * DMODEL;
    for (int d = lane; d < DMODEL; d += 32) {
        const float xv = xr[d];
#pragma unroll
        for (int e = 0; e < NUM_EXPERTS; ++e) acc[e] += xv * gw[e * DMODEL + d];
    }
#pragma unroll
    for (int e = 0; e < NUM_EXPERTS; ++e) {
#pragma unroll
        for (int s = 16; s > 0; s >>= 1) acc[e] += __shfl_xor(acc[e], s, 32);
    }
    if (lane == 0) {
        float g[NUM_EXPERTS];
#pragma unroll
        for (int e = 0; e < NUM_EXPERTS; ++e) g[e] = acc[e] + gb[e];
        int i1 = 0;
#pragma unroll
        for (int e = 1; e < NUM_EXPERTS; ++e) if (g[e] > g[i1]) i1 = e;
        int i2 = (i1 == 0) ? 1 : 0;
#pragma unroll
        for (int e = 0; e < NUM_EXPERTS; ++e)
            if (e != i1 && g[e] > g[i2]) i2 = e;
        const float p1 = 1.0f / (1.0f + expf(g[i2] - g[i1]));
        const float p2 = 1.0f - p1;
        int pos = atomicAdd(&counts[i1], 1);
        tok_id[i1 * TTOK + pos] = t;
        tok_w[i1 * TTOK + pos] = p1;
        pos = atomicAdd(&counts[i2], 1);
        tok_id[i2 * TTOK + pos] = t;
        tok_w[i2 * TTOK + pos] = p2;
    }
}

// ---------- Kernel 2: exclusive prefix over expert counts ----------
__global__ void scan_kernel(const int* __restrict__ counts, int* __restrict__ offsets) {
    if (threadIdx.x == 0) {
        int s = 0;
#pragma unroll
        for (int e = 0; e < NUM_EXPERTS; ++e) { offsets[e] = s; s += counts[e]; }
    }
}

// ---------- Kernel 3: h = gelu(x_gathered @ w1[e] + b1[e]) -> f16 ----------
// Block tile 128x128, 8 waves (4M x 2N), wave tile 32x64, K-step 32.
// Copy ownership: A: thread owns 16 consecutive k of one row (4x b128 loads,
// 2x b128 LDS stores). B: thread owns 2 rows x 8 n (4x b128 loads, transposed
// b16 stores). Register-staged pipeline, last K iteration peeled.
__global__ __launch_bounds__(256) void gemm1_kernel(
    const float* __restrict__ x, const float* __restrict__ w1,
    const float* __restrict__ b1, const int* __restrict__ counts,
    const int* __restrict__ offsets, const int* __restrict__ tok_id,
    _Float16* __restrict__ hbuf) {
    const int e = blockIdx.z;
    const int cnt = counts[e];
    const int mBase = blockIdx.x * TM;
    if (mBase >= cnt) return;
    const int nBase = blockIdx.y * TN;

    __shared__ __align__(16) _Float16 As[TM][LDP];   // [m][k]
    __shared__ __align__(16) _Float16 Bs[TN][LDP];   // transposed [n][k]

    const int tid = threadIdx.x;
    const int lane = tid & 31;
    const int wave = tid >> 5;
    const int waveM = wave >> 1;  // 0..3
    const int waveN = wave & 1;   // 0..1

    // ---- A copy ownership: row = tid>>1, k segment = (tid&1)*16
    const int aRow = tid >> 1;
    const int aK = (tid & 1) * 16;
    int rowC = mBase + aRow;
    rowC = (rowC < cnt - 1) ? rowC : (cnt - 1);          // always-valid gather
    const int myTok = tok_id[e * TTOK + rowC];
    const float* xRow = x + (size_t)myTok * DMODEL + aK;

    // ---- B copy ownership: k rows {bK, bK+16}, n segment = bN..bN+7
    const int bK = tid >> 4;        // 0..15
    const int bN = (tid & 15) * 8;  // 0..120
    const float* bPtr = w1 + (size_t)e * DMODEL * HDIM + (size_t)bK * HDIM + nBase + bN;

    const v8f vzero = {0.f, 0.f, 0.f, 0.f, 0.f, 0.f, 0.f, 0.f};
    v8f acc[2][4];
#pragma unroll
    for (int i = 0; i < 2; ++i)
#pragma unroll
        for (int j = 0; j < 4; ++j) acc[i][j] = vzero;

    float aReg[16], bReg[16];

    auto fetchA = [&](int kk) {
#pragma unroll
        for (int q = 0; q < 4; ++q) {
            const float4 f = *(const float4*)(xRow + kk + q * 4);
            aReg[q * 4 + 0] = f.x; aReg[q * 4 + 1] = f.y;
            aReg[q * 4 + 2] = f.z; aReg[q * 4 + 3] = f.w;
        }
    };
    auto fetchB = [&](int kk) {
#pragma unroll
        for (int it = 0; it < 2; ++it)
#pragma unroll
            for (int q = 0; q < 2; ++q) {
                const float4 f = *(const float4*)(bPtr + (size_t)(kk + it * 16) * HDIM + q * 4);
                bReg[it * 8 + q * 4 + 0] = f.x; bReg[it * 8 + q * 4 + 1] = f.y;
                bReg[it * 8 + q * 4 + 2] = f.z; bReg[it * 8 + q * 4 + 3] = f.w;
            }
    };
    auto stage = [&]() {
        v8h h0, h1;
#pragma unroll
        for (int q = 0; q < 8; ++q) {
            h0[q] = (_Float16)aReg[q];
            h1[q] = (_Float16)aReg[8 + q];
        }
        *(v8h*)&As[aRow][aK] = h0;
        *(v8h*)&As[aRow][aK + 8] = h1;
#pragma unroll
        for (int it = 0; it < 2; ++it)
#pragma unroll
            for (int i = 0; i < 8; ++i)
                Bs[bN + i][bK + it * 16] = (_Float16)bReg[it * 8 + i];
    };
    auto compute = [&]() {
        v16h aF[2], bF[4];
#pragma unroll
        for (int i = 0; i < 2; ++i)
            aF[i] = load_a_frag(&As[waveM * 32 + i * 16][0], lane);
#pragma unroll
        for (int j = 0; j < 4; ++j)
            bF[j] = load_b_frag(&Bs[waveN * 64 + j * 16][0], lane);
#pragma unroll
        for (int i = 0; i < 2; ++i)
#pragma unroll
            for (int j = 0; j < 4; ++j)
                acc[i][j] = wmma_f16(aF[i], bF[j], acc[i][j]);
    };

    fetchA(0); fetchB(0);
    for (int k0 = 0; k0 < DMODEL - TK; k0 += TK) {
        stage();
        __syncthreads();
        fetchA(k0 + TK); fetchB(k0 + TK);   // overlap with WMMA stage
        compute();
        __syncthreads();
    }
    stage();
    __syncthreads();
    compute();

    // epilogue: bias + exact GELU, store f16 into expert-packed h buffer
    const int off = offsets[e];
    const int hi = lane >> 4;
    const int nl = lane & 15;
#pragma unroll
    for (int i = 0; i < 2; ++i) {
#pragma unroll
        for (int j = 0; j < 4; ++j) {
            const int nSub = nBase + waveN * 64 + j * 16 + nl;
            const float bias = b1[e * HDIM + nSub];
#pragma unroll
            for (int r = 0; r < 8; ++r) {
                const int m = mBase + waveM * 32 + i * 16 + hi * 8 + r;
                if (m < cnt) {
                    const float vv = acc[i][j][r] + bias;
                    const float g = 0.5f * vv * (1.0f + erff(vv * 0.70710678f));
                    hbuf[(size_t)(off + m) * HDIM + nSub] = (_Float16)g;
                }
            }
        }
    }
}

// ---------- Kernel 4: out[tok] += w * (h @ w2[e] + b2[e]) ----------
__global__ __launch_bounds__(256) void gemm2_kernel(
    const _Float16* __restrict__ hbuf, const float* __restrict__ w2,
    const float* __restrict__ b2, const int* __restrict__ counts,
    const int* __restrict__ offsets, const int* __restrict__ tok_id,
    const float* __restrict__ tok_w, float* __restrict__ out) {
    const int e = blockIdx.z;
    const int cnt = counts[e];
    const int mBase = blockIdx.x * TM;
    if (mBase >= cnt) return;
    const int nBase = blockIdx.y * TN;

    __shared__ __align__(16) _Float16 As[TM][LDP];
    __shared__ __align__(16) _Float16 Bs[TN][LDP];   // transposed [n][k]

    const int tid = threadIdx.x;
    const int lane = tid & 31;
    const int wave = tid >> 5;
    const int waveM = wave >> 1;
    const int waveN = wave & 1;
    const int off = offsets[e];

    // ---- A: f16 rows straight from packed h buffer (no cvt)
    const int aRow = tid >> 1;
    const int aK = (tid & 1) * 16;
    int rowC = mBase + aRow;
    rowC = (rowC < cnt - 1) ? rowC : (cnt - 1);
    const _Float16* hRow = hbuf + (size_t)(off + rowC) * HDIM + aK;

    const int bK = tid >> 4;
    const int bN = (tid & 15) * 8;
    const float* bPtr = w2 + (size_t)e * HDIM * DMODEL + (size_t)bK * DMODEL + nBase + bN;

    const v8f vzero = {0.f, 0.f, 0.f, 0.f, 0.f, 0.f, 0.f, 0.f};
    v8f acc[2][4];
#pragma unroll
    for (int i = 0; i < 2; ++i)
#pragma unroll
        for (int j = 0; j < 4; ++j) acc[i][j] = vzero;

    v8h aH[2];
    float bReg[16];

    auto fetchA = [&](int kk) {
        aH[0] = *(const v8h*)(hRow + kk);
        aH[1] = *(const v8h*)(hRow + kk + 8);
    };
    auto fetchB = [&](int kk) {
#pragma unroll
        for (int it = 0; it < 2; ++it)
#pragma unroll
            for (int q = 0; q < 2; ++q) {
                const float4 f = *(const float4*)(bPtr + (size_t)(kk + it * 16) * DMODEL + q * 4);
                bReg[it * 8 + q * 4 + 0] = f.x; bReg[it * 8 + q * 4 + 1] = f.y;
                bReg[it * 8 + q * 4 + 2] = f.z; bReg[it * 8 + q * 4 + 3] = f.w;
            }
    };
    auto stage = [&]() {
        *(v8h*)&As[aRow][aK] = aH[0];
        *(v8h*)&As[aRow][aK + 8] = aH[1];
#pragma unroll
        for (int it = 0; it < 2; ++it)
#pragma unroll
            for (int i = 0; i < 8; ++i)
                Bs[bN + i][bK + it * 16] = (_Float16)bReg[it * 8 + i];
    };
    auto compute = [&]() {
        v16h aF[2], bF[4];
#pragma unroll
        for (int i = 0; i < 2; ++i)
            aF[i] = load_a_frag(&As[waveM * 32 + i * 16][0], lane);
#pragma unroll
        for (int j = 0; j < 4; ++j)
            bF[j] = load_b_frag(&Bs[waveN * 64 + j * 16][0], lane);
#pragma unroll
        for (int i = 0; i < 2; ++i)
#pragma unroll
            for (int j = 0; j < 4; ++j)
                acc[i][j] = wmma_f16(aF[i], bF[j], acc[i][j]);
    };

    fetchA(0); fetchB(0);
    for (int k0 = 0; k0 < HDIM - TK; k0 += TK) {
        stage();
        __syncthreads();
        fetchA(k0 + TK); fetchB(k0 + TK);
        compute();
        __syncthreads();
    }
    stage();
    __syncthreads();
    compute();

    const int hi = lane >> 4;
    const int nl = lane & 15;
#pragma unroll
    for (int i = 0; i < 2; ++i) {
#pragma unroll
        for (int j = 0; j < 4; ++j) {
            const int nSub = nBase + waveN * 64 + j * 16 + nl;
            const float bias = b2[e * DMODEL + nSub];
#pragma unroll
            for (int r = 0; r < 8; ++r) {
                const int m = mBase + waveM * 32 + i * 16 + hi * 8 + r;
                if (m < cnt) {
                    const int tkn = tok_id[e * TTOK + m];
                    const float wgt = tok_w[e * TTOK + m];
                    atomicAdd(&out[(size_t)tkn * DMODEL + nSub],
                              wgt * (acc[i][j][r] + bias));
                }
            }
        }
    }
}

extern "C" void kernel_launch(void* const* d_in, const int* in_sizes, int n_in,
                              void* d_out, int out_size, void* d_ws, size_t ws_size,
                              hipStream_t stream) {
    const float* x      = (const float*)d_in[0];
    const float* gate_w = (const float*)d_in[1];
    const float* gate_b = (const float*)d_in[2];
    const float* w1     = (const float*)d_in[3];
    const float* b1     = (const float*)d_in[4];
    const float* w2     = (const float*)d_in[5];
    const float* b2     = (const float*)d_in[6];
    float* out = (float*)d_out;

    char* ws = (char*)d_ws;
    int* counts  = (int*)ws;                                  // 8 ints
    int* offsets = counts + NUM_EXPERTS;                      // 8 ints
    int* tok_id  = offsets + NUM_EXPERTS;                     // E*T ints
    float* tok_w = (float*)(tok_id + NUM_EXPERTS * TTOK);     // E*T floats
    _Float16* hbuf = (_Float16*)(tok_w + NUM_EXPERTS * TTOK); // 2T*H halves (~134 MB)

    hipMemsetAsync(counts, 0, NUM_EXPERTS * sizeof(int), stream);
    hipMemsetAsync(d_out, 0, (size_t)out_size * sizeof(float), stream);

    gate_kernel<<<TTOK / 8, 256, 0, stream>>>(x, gate_w, gate_b, counts, tok_id, tok_w);
    scan_kernel<<<1, 32, 0, stream>>>(counts, offsets);

    dim3 g1(TTOK / TM, HDIM / TN, NUM_EXPERTS);
    gemm1_kernel<<<g1, 256, 0, stream>>>(x, w1, b1, counts, offsets, tok_id, hbuf);

    dim3 g2(TTOK / TM, DMODEL / TN, NUM_EXPERTS);
    gemm2_kernel<<<g2, 256, 0, stream>>>(hbuf, w2, b2, counts, offsets, tok_id, tok_w, out);
}
